// TinySwinTransformer_38096359915897
// MI455X (gfx1250) — matrix operand order repaired and checked by reading, not compiled, not run on hardware
//
#include <hip/hip_runtime.h>
#include <hip/hip_bf16.h>
#include <cstdint>
#include <cstddef>

typedef __attribute__((ext_vector_type(8)))  float  v8f;
typedef __attribute__((ext_vector_type(16))) __bf16 v16bf;
typedef __attribute__((ext_vector_type(4)))  float  f4;
typedef __attribute__((ext_vector_type(2)))  float  f2;

static __device__ __forceinline__ __bf16 tobf(float f) {
  return static_cast<__bf16>(f);   // RNE, hardware cvt on gfx1250
}

static __device__ __forceinline__ v8f wmma_bf16(v16bf a, v16bf b, v8f c) {
  return __builtin_amdgcn_wmma_f32_16x16x32_bf16(false, a, false, b, (short)0, c, false, false);
}

// half-wave (16-lane) row reductions: xor offsets {1,2,4,8} stay inside a 16-lane group
static __device__ __forceinline__ float halfmax(float v) {
  v = fmaxf(v, __shfl_xor(v, 1)); v = fmaxf(v, __shfl_xor(v, 2));
  v = fmaxf(v, __shfl_xor(v, 4)); v = fmaxf(v, __shfl_xor(v, 8));
  return v;
}
static __device__ __forceinline__ float halfsum(float v) {
  v += __shfl_xor(v, 1); v += __shfl_xor(v, 2);
  v += __shfl_xor(v, 4); v += __shfl_xor(v, 8);
  return v;
}

// Fill an A-layout fragment (16x32 bf16) from a row pointer, K-step base kk.
// Lane l15 = row; 16 contiguous K values starting at kk + 16*half.
static __device__ __forceinline__ v16bf load_afrag(const float* __restrict__ row,
                                                   int kk, int half) {
  const f4* r4 = reinterpret_cast<const f4*>(row);
  int a0 = (kk + (half << 4)) >> 2;
  int a2 = (kk + 8 + (half << 4)) >> 2;
  f4 ra0 = r4[a0], ra1 = r4[a0 + 1], ra2 = r4[a2], ra3 = r4[a2 + 1];
  v16bf af;
  #pragma unroll
  for (int t = 0; t < 4; ++t) {
    af[t]      = tobf(ra0[t]);
    af[4 + t]  = tobf(ra1[t]);
    af[8 + t]  = tobf(ra2[t]);
    af[12 + t] = tobf(ra3[t]);
  }
  return af;
}

// Fill a B-layout fragment from a row pointer (weight row = output column):
// 16 contiguous K values starting at kk + 16*half.
static __device__ __forceinline__ v16bf load_bfrag(const float* __restrict__ row,
                                                   int kk, int half) {
  const f4* r4 = reinterpret_cast<const f4*>(row);
  int b0 = (kk + (half << 4)) >> 2;
  f4 rb0 = r4[b0], rb1 = r4[b0 + 1], rb2 = r4[b0 + 2], rb3 = r4[b0 + 3];
  v16bf bf;
  #pragma unroll
  for (int t = 0; t < 4; ++t) {
    bf[t]      = tobf(rb0[t]);
    bf[4 + t]  = tobf(rb1[t]);
    bf[8 + t]  = tobf(rb2[t]);
    bf[12 + t] = tobf(rb3[t]);
  }
  return bf;
}

// ---------------------------------------------------------------------------
// Kernel 1: depth mean (8) + bilinear 2x downsample (== 2x2 average) fused.
// x[4,256,8,128,128] -> X0[b, token(oy*64+ox), c]  (token-major for GEMM)
// ---------------------------------------------------------------------------
__global__ void pool_kernel(const float* __restrict__ x, float* __restrict__ X0) {
  size_t idx = (size_t)blockIdx.x * blockDim.x + threadIdx.x;
  if (idx >= (size_t)4 * 256 * 64 * 64) return;
  int ox = idx & 63;
  int oy = (idx >> 6) & 63;
  int c  = (idx >> 12) & 255;
  int bb = (int)(idx >> 20);
  const float* base = x + ((size_t)bb * 256 + c) * 8 * 128 * 128 + (size_t)(oy * 2) * 128;
  float s = 0.f;
  #pragma unroll
  for (int d = 0; d < 8; ++d) {
    const f2* p2 = reinterpret_cast<const f2*>(base + (size_t)d * 128 * 128);
    f2 a = p2[ox], b = p2[ox + 64];
    s += (a[0] + a[1]) + (b[0] + b[1]);
  }
  X0[((size_t)bb * 4096 + oy * 64 + ox) * 256 + c] = s * (1.f / 32.f);
}

// ---------------------------------------------------------------------------
// Kernel 2: WMMA GEMM  out = A @ W^T (+bias) (+residual)
// A:[M,K] f32 row-major, W:[N,K] f32 row-major, out:[M,N].
// One wave per 16x32 output tile (A fragment reused across two WMMAs).
// All M %16==0, N %32==0, K %32==0 in this model.
// ---------------------------------------------------------------------------
__global__ void gemm_wmma(const float* __restrict__ A, const float* __restrict__ W,
                          const float* __restrict__ bias, const float* __restrict__ res,
                          float* __restrict__ out, int M, int N, int K) {
  int wave = blockIdx.x * (blockDim.x >> 5) + (threadIdx.x >> 5);
  int lane = threadIdx.x & 31;
  int tilesN = N >> 5;
  int tiles = (M >> 4) * tilesN;
  if (wave >= tiles) return;
  int tm = wave / tilesN, tn = wave - tm * tilesN;
  int row0 = tm << 4, col0 = tn << 5;
  int half = lane >> 4, l15 = lane & 15;

  const float* arow  = A + (size_t)(row0 + l15) * K;
  const float* wrow0 = W + (size_t)(col0 + l15) * K;
  const float* wrow1 = wrow0 + (size_t)16 * K;
  v8f acc0 = {0.f, 0.f, 0.f, 0.f, 0.f, 0.f, 0.f, 0.f};
  v8f acc1 = acc0;

  for (int kk = 0; kk < K; kk += 32) {
    v16bf af = load_afrag(arow, kk, half);
    v16bf b0 = load_bfrag(wrow0, kk, half);
    v16bf b1 = load_bfrag(wrow1, kk, half);
    acc0 = wmma_bf16(af, b0, acc0);
    acc1 = wmma_bf16(af, b1, acc1);
  }

  int n0 = col0 + l15, n1 = n0 + 16;
  float bias0 = bias ? bias[n0] : 0.f;
  float bias1 = bias ? bias[n1] : 0.f;
  int mbase = row0 + (half << 3);
  if (res) {
    #pragma unroll
    for (int i = 0; i < 8; ++i) {
      size_t r = (size_t)(mbase + i) * N;
      out[r + n0] = acc0[i] + bias0 + res[r + n0];
      out[r + n1] = acc1[i] + bias1 + res[r + n1];
    }
  } else {
    #pragma unroll
    for (int i = 0; i < 8; ++i) {
      size_t r = (size_t)(mbase + i) * N;
      out[r + n0] = acc0[i] + bias0;
      out[r + n1] = acc1[i] + bias1;
    }
  }
}

// ---------------------------------------------------------------------------
// Kernel 3: LayerNorm, one wave per row
// ---------------------------------------------------------------------------
__global__ void ln_kernel(const float* __restrict__ in, const float* __restrict__ g,
                          const float* __restrict__ b, float* __restrict__ out,
                          int M, int C) {
  int row = blockIdx.x * (blockDim.x >> 5) + (threadIdx.x >> 5);
  int lane = threadIdx.x & 31;
  if (row >= M) return;
  const float* r = in + (size_t)row * C;
  float s = 0.f, ss = 0.f;
  for (int c = lane; c < C; c += 32) { float v = r[c]; s += v; ss += v * v; }
  for (int off = 16; off; off >>= 1) { s += __shfl_xor(s, off); ss += __shfl_xor(ss, off); }
  float mean = s / C;
  float var = ss / C - mean * mean;
  float rs = rsqrtf(var + 1e-5f);
  float* o = out + (size_t)row * C;
  for (int c = lane; c < C; c += 32) o[c] = (r[c] - mean) * rs * g[c] + b[c];
}

// ---------------------------------------------------------------------------
// Kernel 4: flash attention with relative-position bias. One wave (block of
// 32) per (b, h, 16-query tile). qkv:[B*L, 3C] with per-row layout
// [q(h,hd) | k(h,hd) | v(h,hd)], hd=32. out:[B*L, C] (pre-proj attention out).
// ---------------------------------------------------------------------------
__global__ void flash_kernel(const float* __restrict__ qkv, const float* __restrict__ rpb,
                             float* __restrict__ out, int Ltok, int C, int heads,
                             int win, int shift, int maskv) {
  const int hd = 32;
  int ntiles = Ltok >> 4;
  int qt   = blockIdx.x % ntiles;
  int h    = (blockIdx.x / ntiles) % heads;
  int bb   = blockIdx.x / (ntiles * heads);
  int qbase = qt << 4;
  int lane = threadIdx.x & 31;
  int half = lane >> 4, l15 = lane & 15;
  const float scale = 0.17677669529663687f;  // 32^-0.5
  const size_t rstride = 3 * (size_t)C;
  const int twin = 2 * win - 1;

  // Q fragment (A layout, 16x32); scale folded in after vector load
  const float* Qp = qkv + ((size_t)bb * Ltok + qbase + l15) * rstride + h * hd;
  v16bf qa;
  {
    const f4* q4 = reinterpret_cast<const f4*>(Qp);
    int a0 = (half << 4) >> 2, a2 = (8 + (half << 4)) >> 2;
    f4 r0 = q4[a0], r1 = q4[a0 + 1], r2 = q4[a2], r3 = q4[a2 + 1];
    #pragma unroll
    for (int t = 0; t < 4; ++t) {
      qa[t]      = tobf(r0[t] * scale);
      qa[4 + t]  = tobf(r1[t] * scale);
      qa[8 + t]  = tobf(r2[t] * scale);
      qa[12 + t] = tobf(r3[t] * scale);
    }
  }

  float mi[8], li[8];
  v8f O0 = {0.f,0.f,0.f,0.f,0.f,0.f,0.f,0.f};
  v8f O1 = {0.f,0.f,0.f,0.f,0.f,0.f,0.f,0.f};
  int qrow[8];
  #pragma unroll
  for (int i = 0; i < 8; ++i) { mi[i] = -1e30f; li[i] = 0.f; qrow[i] = qbase + i + (half << 3); }

  __shared__ __bf16 Pl[16][32];

  for (int kb = 0; kb < Ltok; kb += 32) {
    // K fragments (B layout): columns = keys, K-dim = hd; 16 contiguous floats
    const float* Kp0 = qkv + ((size_t)bb * Ltok + kb + l15) * rstride + C + h * hd + (half << 4);
    const float* Kp1 = Kp0 + 16 * rstride;
    v16bf kf0, kf1;
    {
      const f4* k04 = reinterpret_cast<const f4*>(Kp0);
      const f4* k14 = reinterpret_cast<const f4*>(Kp1);
      f4 a0 = k04[0], a1 = k04[1], a2 = k04[2], a3 = k04[3];
      f4 c0 = k14[0], c1 = k14[1], c2 = k14[2], c3 = k14[3];
      #pragma unroll
      for (int t = 0; t < 4; ++t) {
        kf0[t] = tobf(a0[t]); kf0[4 + t] = tobf(a1[t]);
        kf0[8 + t] = tobf(a2[t]); kf0[12 + t] = tobf(a3[t]);
        kf1[t] = tobf(c0[t]); kf1[4 + t] = tobf(c1[t]);
        kf1[8 + t] = tobf(c2[t]); kf1[12 + t] = tobf(c3[t]);
      }
    }
    v8f z0 = {0.f,0.f,0.f,0.f,0.f,0.f,0.f,0.f};
    v8f s0 = wmma_bf16(qa, kf0, z0);
    v8f s1 = wmma_bf16(qa, kf1, z0);

    // bias + streaming softmax (per row i; row lives on 16 lanes of a half)
    #pragma unroll
    for (int i = 0; i < 8; ++i) {
      int q = qrow[i];
      int k0 = kb + l15, k1 = k0 + 16;
      int dy0 = (q >> shift) - (k0 >> shift), dx0 = (q & maskv) - (k0 & maskv);
      int dy1 = (q >> shift) - (k1 >> shift), dx1 = (q & maskv) - (k1 & maskv);
      float b0 = rpb[((dy0 + win - 1) * twin + dx0 + win - 1) * heads + h];
      float b1 = rpb[((dy1 + win - 1) * twin + dx1 + win - 1) * heads + h];
      float v0 = s0[i] + b0, v1 = s1[i] + b1;
      float rmax = halfmax(fmaxf(v0, v1));
      float mnew = fmaxf(mi[i], rmax);
      float alpha = __expf(mi[i] - mnew);
      O0[i] *= alpha; O1[i] *= alpha; li[i] *= alpha; mi[i] = mnew;
      float e0 = __expf(v0 - mnew), e1 = __expf(v1 - mnew);
      li[i] += halfsum(e0 + e1);
      Pl[i + (half << 3)][l15]      = tobf(e0);
      Pl[i + (half << 3)][l15 + 16] = tobf(e1);
    }
    __syncthreads();

    // P as A fragment (16x32) from LDS
    v16bf pa;
    #pragma unroll
    for (int j = 0; j < 16; ++j) {
      int k = j + (half << 4);
      pa[j] = Pl[l15][k];
    }
    // V fragments (B layout): columns = hd dims, K-dim = 32 keys of this block
    const float* Vp = qkv + ((size_t)bb * Ltok + kb + (half << 4)) * rstride + 2 * C + h * hd;
    v16bf vf0, vf1;
    #pragma unroll
    for (int j = 0; j < 16; ++j) {
      const float* vr = Vp + (size_t)j * rstride;
      vf0[j] = tobf(vr[l15]);
      vf1[j] = tobf(vr[l15 + 16]);
    }
    O0 = wmma_bf16(pa, vf0, O0);
    O1 = wmma_bf16(pa, vf1, O1);
    __syncthreads();
  }

  #pragma unroll
  for (int i = 0; i < 8; ++i) {
    int q = qrow[i];
    float inv = 1.f / li[i];
    float* op = out + ((size_t)bb * Ltok + q) * C + h * hd;
    op[l15]      = O0[i] * inv;
    op[l15 + 16] = O1[i] * inv;
  }
}

// ---------------------------------------------------------------------------
// Kernel 5: exact GELU, in place
// ---------------------------------------------------------------------------
__global__ void gelu_kernel(float* __restrict__ x, size_t n) {
  size_t i = (size_t)blockIdx.x * blockDim.x + threadIdx.x;
  if (i < n) {
    float v = x[i];
    x[i] = 0.5f * v * (1.f + erff(v * 0.70710678118654752f));
  }
}

// ---------------------------------------------------------------------------
// Kernel 6: patch merge  t[4,4096,96] -> out[4,1024,384]
// channel quadrants: [x(0::2,0::2) | x(1::2,0::2) | x(0::2,1::2) | x(1::2,1::2)]
// ---------------------------------------------------------------------------
__global__ void merge_kernel(const float* __restrict__ t, float* __restrict__ out) {
  size_t idx = (size_t)blockIdx.x * blockDim.x + threadIdx.x;
  if (idx >= (size_t)4 * 1024 * 384) return;
  int c   = idx % 384;
  int tok = (int)((idx / 384) % 1024);
  int bb  = (int)(idx / (384 * 1024));
  int i = tok >> 5, j = tok & 31;
  int quad = c / 96, cc = c - quad * 96;
  int sy = 2 * i + (quad & 1);
  int sx = 2 * j + (quad >> 1);
  out[idx] = t[((size_t)bb * 4096 + sy * 64 + sx) * 96 + cc];
}

// ---------------------------------------------------------------------------
// Kernel 7: mean over tokens: in[4, M, C] -> out[4, C]
// ---------------------------------------------------------------------------
__global__ void mean_kernel(const float* __restrict__ in, float* __restrict__ out,
                            int M, int C) {
  int idx = blockIdx.x * blockDim.x + threadIdx.x;
  if (idx >= 4 * C) return;
  int bb = idx / C, c = idx - bb * C;
  float s = 0.f;
  for (int l = 0; l < M; ++l) s += in[((size_t)bb * M + l) * C + c];
  out[idx] = s / M;
}

// ---------------------------------------------------------------------------
struct BlockP {
  const float *ln1g, *ln1b, *qkvw, *qkvb, *rpb, *projw, *projb,
              *ln2g, *ln2b, *fc1w, *fc1b, *fc2w, *fc2b;
};

extern "C" void kernel_launch(void* const* d_in, const int* in_sizes, int n_in,
                              void* d_out, int out_size, void* d_ws, size_t ws_size,
                              hipStream_t stream) {
  (void)in_sizes; (void)n_in; (void)out_size; (void)ws_size;
  int pi = 0;
  auto nxt = [&]() { return (const float*)d_in[pi++]; };

  const float* x     = nxt();
  const float* pe_w  = nxt();
  const float* pe_b  = nxt();
  const float* pe_ng = nxt();
  const float* pe_nb = nxt();
  auto readBlock = [&]() {
    BlockP p;
    p.ln1g = nxt(); p.ln1b = nxt(); p.qkvw = nxt(); p.qkvb = nxt(); p.rpb = nxt();
    p.projw = nxt(); p.projb = nxt(); p.ln2g = nxt(); p.ln2b = nxt();
    p.fc1w = nxt(); p.fc1b = nxt(); p.fc2w = nxt(); p.fc2b = nxt();
    return p;
  };
  BlockP s0b0 = readBlock();
  BlockP s0b1 = readBlock();
  const float* mg_g  = nxt();
  const float* mg_b  = nxt();
  const float* red_w = nxt();
  BlockP s1b0 = readBlock();
  BlockP s1b1 = readBlock();
  const float* fn_g = nxt();
  const float* fn_b = nxt();

  // workspace (floats). X0 overlays qkvBuf: X0 dead before first qkv GEMM.
  float* ws = (float*)d_ws;
  size_t off = 0;
  auto alloc = [&](size_t n) { float* p = ws + off; off += n; return p; };
  float* qkvBuf = alloc(16384ull * 288);  // 4,718,592 (>= X0's 4,194,304)
  float* X0     = qkvBuf;
  float* xA     = alloc(16384ull * 96);
  float* xB     = alloc(16384ull * 96);
  float* yBuf   = alloc(16384ull * 96);   // also merge-LN out (4096*384, same size)
  float* attBuf = alloc(16384ull * 96);
  float* hBuf   = alloc(16384ull * 384);  // MLP hidden / merge staging

  auto gemm = [&](const float* A, const float* W, const float* bias, const float* res,
                  float* out, int M, int N, int K) {
    int tiles = (M >> 4) * (N >> 5);    // 16x32 tiles per wave
    gemm_wmma<<<(tiles + 3) / 4, 128, 0, stream>>>(A, W, bias, res, out, M, N, K);
  };
  auto ln = [&](const float* in, const float* g, const float* b, float* out, int M, int C) {
    ln_kernel<<<(M + 3) / 4, 128, 0, stream>>>(in, g, b, out, M, C);
  };
  auto runBlock = [&](const BlockP& p, const float* xin, float* xout,
                      int Bn, int Ltok, int C, int heads, int win, int shift) {
    int M = Bn * Ltok;
    ln(xin, p.ln1g, p.ln1b, yBuf, M, C);
    gemm(yBuf, p.qkvw, p.qkvb, nullptr, qkvBuf, M, 3 * C, C);
    flash_kernel<<<Bn * heads * (Ltok >> 4), 32, 0, stream>>>(
        qkvBuf, p.rpb, attBuf, Ltok, C, heads, win, shift, win - 1);
    gemm(attBuf, p.projw, p.projb, xin, xout, M, C, C);   // x = x + proj(attn)
    ln(xout, p.ln2g, p.ln2b, yBuf, M, C);
    gemm(yBuf, p.fc1w, p.fc1b, nullptr, hBuf, M, 4 * C, C);
    size_t hn = (size_t)M * 4 * C;
    gelu_kernel<<<(unsigned)((hn + 255) / 256), 256, 0, stream>>>(hBuf, hn);
    gemm(hBuf, p.fc2w, p.fc2b, xout, xout, M, C, 4 * C);  // x += mlp (in-place residual)
  };

  // 1. input reduction: depth mean + 2x2 avg (bilinear /2), token-major output
  pool_kernel<<<(4 * 256 * 64 * 64 + 255) / 256, 256, 0, stream>>>(x, X0);
  // 2. patch embed: [16384,256] @ [96,256]^T + bias, then LN
  gemm(X0, pe_w, pe_b, nullptr, xB, 16384, 96, 256);
  ln(xB, pe_ng, pe_nb, xA, 16384, 96);
  // 3. stage 0: L=4096, C=96, h=3, window 64
  runBlock(s0b0, xA, xB, 4, 4096, 96, 3, 64, 6);
  runBlock(s0b1, xB, xA, 4, 4096, 96, 3, 64, 6);
  // 4. patch merging -> LN(384) -> Linear(384->192, no bias)
  merge_kernel<<<(4 * 1024 * 384 + 255) / 256, 256, 0, stream>>>(xA, hBuf);
  ln(hBuf, mg_g, mg_b, yBuf, 4096, 384);
  gemm(yBuf, red_w, nullptr, nullptr, xB, 4096, 192, 384);
  // 5. stage 1: L=1024, C=192, h=6, window 32
  runBlock(s1b0, xB, xA, 4, 1024, 192, 6, 32, 5);
  runBlock(s1b1, xA, xB, 4, 1024, 192, 6, 32, 5);
  // 6. final LN + token mean -> [4,192]
  ln(xB, fn_g, fn_b, yBuf, 4096, 192);
  mean_kernel<<<(768 + 255) / 256, 256, 0, stream>>>(yBuf, (float*)d_out, 1024, 192);
}
